// tempolMix_CI_pat_78417512890551
// MI455X (gfx1250) — matrix operand order repaired and checked by reading, not compiled, not run on hardware
//
#include <hip/hip_runtime.h>
#include <math.h>

// ---------------------------------------------------------------------------
// Fused per-patch MLP + GELU + residual + LayerNorm for MI455X (gfx1250).
// B=2048, P=54, D=512.  3-term bf16 split-precision WMMA GEMMs (fp32-class
// accuracy at matrix-core rate).  X converted to bf16 hi/lo planes once;
// W tiles staged per K-step with prefetch pipelining; H overlays X; res
// overlays H.  192KB LDS per workgroup.
// ---------------------------------------------------------------------------

typedef __attribute__((ext_vector_type(16))) __bf16 v16bf;
typedef __attribute__((ext_vector_type(8)))  __bf16 v8bf;
typedef __attribute__((ext_vector_type(2)))  __bf16 v2bf;
typedef __attribute__((ext_vector_type(8)))  float  v8f;

#define NP 54
#define ND 512
#define BM 64           // rows per workgroup
#define KT 32           // K-tile per step
#define NSTEP (ND / KT) // 16

// LDS layout (u16 units):
//  Ahi [64x512] @ 0      (32768 u16)  X planes -> H planes -> f32 res buffer
//  Alo [64x512] @ 32768
//  Whi [512x32] @ 65536  (16384 u16)
//  Wlo [512x32] @ 81920
// total = 98304 u16 = 196608 bytes (192KB of the WGP's 320KB)
#define LDS_BYTES 196608

#ifdef __has_builtin
#if __has_builtin(__builtin_amdgcn_cvt_pk_bf16_f32)
#define HAVE_CVT_PK_BF16 1
#endif
#endif
#ifndef HAVE_CVT_PK_BF16
#define HAVE_CVT_PK_BF16 0
#endif

__device__ __forceinline__ unsigned short f32_to_bf16_rne(float f) {
  unsigned int u = __float_as_uint(f);
  unsigned int r = u + 0x7FFFu + ((u >> 16) & 1u);
  return (unsigned short)(r >> 16);
}
__device__ __forceinline__ float bf16_to_f32(unsigned short h) {
  return __uint_as_float(((unsigned int)h) << 16);
}

// Pack two f32 -> two bf16 (RNE) in one u32: [15:0]=a, [31:16]=b.
__device__ __forceinline__ unsigned int pack_bf16x2(float a, float b) {
#if HAVE_CVT_PK_BF16
  v2bf v = __builtin_amdgcn_cvt_pk_bf16_f32(a, b);
  return __builtin_bit_cast(unsigned int, v);
#else
  return ((unsigned int)f32_to_bf16_rne(b) << 16) | (unsigned int)f32_to_bf16_rne(a);
#endif
}

// Split-convert 4 f32 into bf16 hi/lo planes (hi + lo ~= f32 to ~2^-16 rel).
__device__ __forceinline__ void split_store4(unsigned short* hi,
                                             unsigned short* lo,
                                             int idx, float4 v) {
  unsigned int h01 = pack_bf16x2(v.x, v.y);
  unsigned int h23 = pack_bf16x2(v.z, v.w);
  float r0 = v.x - __uint_as_float(h01 << 16);
  float r1 = v.y - __uint_as_float(h01 & 0xFFFF0000u);
  float r2 = v.z - __uint_as_float(h23 << 16);
  float r3 = v.w - __uint_as_float(h23 & 0xFFFF0000u);
  uint2 hv, lv;
  hv.x = h01; hv.y = h23;
  lv.x = pack_bf16x2(r0, r1);
  lv.y = pack_bf16x2(r2, r3);
  *(uint2*)(hi + idx) = hv;
  *(uint2*)(lo + idx) = lv;
}

__device__ __forceinline__ v16bf ld_frag(const unsigned short* base,
                                         int off0, int off1) {
  v8bf a = *(const v8bf*)(base + off0);
  v8bf b = *(const v8bf*)(base + off1);
  return __builtin_shufflevector(a, b, 0, 1, 2, 3, 4, 5, 6, 7,
                                 8, 9, 10, 11, 12, 13, 14, 15);
}

// Stage one [512 x 32] f32 K-slice of W into bf16 hi/lo planes.
// pf (may be null): base of the *next* K-slice, prefetched one step ahead.
__device__ __forceinline__ void stage_W(unsigned short* Whi, unsigned short* Wlo,
                                        const float* __restrict__ Wsrc, int k0,
                                        const float* __restrict__ pf, int tid) {
#pragma unroll
  for (int i = 0; i < 16; ++i) {
    int idx = tid + i * 256;
    int n = idx >> 3;
    int q = (idx & 7) * 4;
    if (pf) __builtin_prefetch(pf + (long)n * ND + q, 0, 0);
    float4 v = *(const float4*)(Wsrc + (long)n * ND + k0 + q);
    split_store4(Whi, Wlo, n * KT + q, v);
  }
}

// One 512-K GEMM phase: acc[MxN tile set] += Asplit @ Wsrc^T (3-term bf16).
// A planes are [64 x 512] bf16 hi/lo already resident in LDS.
__device__ __forceinline__ void gemm_phase(
    v8f acc[16], const unsigned short* Ahi, const unsigned short* Alo,
    unsigned short* Whi, unsigned short* Wlo,
    const float* __restrict__ Wsrc, const float* __restrict__ Wnext,
    int tid, int arow, int cbase, int laneN, int laneHalf) {
#pragma unroll 1
  for (int step = 0; step < NSTEP; ++step) {
    const int k0 = step * KT;
    const float* pf = (step + 1 < NSTEP) ? (Wsrc + (step + 1) * KT) : Wnext;
    __syncthreads();                 // all waves done reading previous W tile
    stage_W(Whi, Wlo, Wsrc, k0, pf, tid);
    __syncthreads();                 // W tile ready

    // A fragments (16x32 bf16): lane<16 -> M=lane, K = 16*laneHalf + 0..15
    const int aoff0 = arow * ND + k0 + laneHalf * 16;
    v16bf Ah = ld_frag(Ahi, aoff0, aoff0 + 8);
    v16bf Al = ld_frag(Alo, aoff0, aoff0 + 8);
    // B fragments (32x16 bf16): col N = lane%16, K = 16*laneHalf + 0..15
    int boff = (cbase + laneN) * KT + laneHalf * 16;
    v16bf Bh = ld_frag(Whi, boff, boff + 8);
    v16bf Bl = ld_frag(Wlo, boff, boff + 8);
#pragma unroll
    for (int t = 0; t < 16; ++t) {
      v16bf Bh_n = Bh, Bl_n = Bl;
      if (t + 1 < 16) {              // one-tile lookahead on B fragments
        int bo = (cbase + (t + 1) * 16 + laneN) * KT + laneHalf * 16;
        Bh_n = ld_frag(Whi, bo, bo + 8);
        Bl_n = ld_frag(Wlo, bo, bo + 8);
      }
      acc[t] = __builtin_amdgcn_wmma_f32_16x16x32_bf16(false, Ah, false, Bh,
                                                       (short)0, acc[t], false, false);
      acc[t] = __builtin_amdgcn_wmma_f32_16x16x32_bf16(false, Al, false, Bh,
                                                       (short)0, acc[t], false, false);
      acc[t] = __builtin_amdgcn_wmma_f32_16x16x32_bf16(false, Ah, false, Bl,
                                                       (short)0, acc[t], false, false);
      Bh = Bh_n; Bl = Bl_n;
    }
  }
}

__global__ __launch_bounds__(256) void fused_patch_mlp_ln_kernel(
    const float* __restrict__ x,  const float* __restrict__ W1,
    const float* __restrict__ b1, const float* __restrict__ W2,
    const float* __restrict__ b2, const float* __restrict__ gamma,
    const float* __restrict__ beta, float* __restrict__ out) {
  extern __shared__ unsigned short sm[];
  unsigned short* Ahi = sm;          // X planes, then H planes
  unsigned short* Alo = sm + 32768;
  unsigned short* Whi = sm + 65536;
  unsigned short* Wlo = sm + 81920;
  float* res = (float*)sm;           // overlays A planes after GEMM2

  const int p  = blockIdx.y;
  const int b0 = blockIdx.x * BM;
  const int tid = threadIdx.x;
  const int wave = tid >> 5;
  const int lane = tid & 31;
  const int laneN = lane & 15;
  const int laneHalf = lane >> 4;
  const int r  = wave & 3;        // 16-row tile index within BM
  const int cg = wave >> 2;       // column half (0 -> cols 0..255, 1 -> 256..511)
  const int cbase = cg * 256;
  const int arow = r * 16 + laneN;

  const float* xblk = x + ((long)b0 * NP + p) * ND;  // row stride NP*ND
  const float* W1p  = W1 + (long)p * ND * ND;
  const float* W2p  = W2 + (long)p * ND * ND;

  v8f zero = {};
  v8f acc[16];
#pragma unroll
  for (int t = 0; t < 16; ++t) acc[t] = zero;

  // -------- Stage full X block [64 x 512] into bf16 hi/lo planes (once) ----
#pragma unroll
  for (int i = 0; i < 32; ++i) {
    int idx = tid + i * 256;         // 8192 float4 chunks
    int row = idx >> 7;
    int q   = (idx & 127) * 4;
    float4 v = *(const float4*)(xblk + (long)row * (NP * ND) + q);
    split_store4(Ahi, Alo, row * ND + q, v);
  }
  // (first barrier inside gemm_phase publishes the X planes)

  // ===================== Phase 1: h = gelu(x @ W1^T + b1) ===================
  gemm_phase(acc, Ahi, Alo, Whi, Wlo, W1p, /*prefetch next*/ W2p,
             tid, arow, cbase, laneN, laneHalf);

  __syncthreads();   // all waves done reading X planes; safe to overwrite w/ H
  // bias + exact-erf GELU, split-store h into (reused) A planes
  const float inv_sqrt2 = 0.70710678118654752f;
#pragma unroll
  for (int t = 0; t < 16; ++t) {
    const int d = cbase + t * 16 + laneN;
    const float bv = b1[p * ND + d];
#pragma unroll
    for (int j = 0; j < 8; ++j) {
      const int m = r * 16 + laneHalf * 8 + j;  // C/D layout: VGPR j -> M=j / M=8+j
      float v = acc[t][j] + bv;
      float g = 0.5f * v * (1.0f + erff(v * inv_sqrt2));
      unsigned short hb = f32_to_bf16_rne(g);
      Ahi[m * ND + d] = hb;
      Alo[m * ND + d] = f32_to_bf16_rne(g - bf16_to_f32(hb));
    }
    acc[t] = zero;
  }

  // ===================== Phase 2: o = h @ W2^T =============================
  gemm_phase(acc, Ahi, Alo, Whi, Wlo, W2p, /*prefetch next*/ nullptr,
             tid, arow, cbase, laneN, laneHalf);

  __syncthreads();
  // dump raw o into f32 res buffer (overlays dead H planes)
#pragma unroll
  for (int t = 0; t < 16; ++t) {
    const int d = cbase + t * 16 + laneN;
#pragma unroll
    for (int j = 0; j < 8; ++j) {
      const int m = r * 16 + laneHalf * 8 + j;
      res[m * ND + d] = acc[t][j];
    }
  }
  __syncthreads();

  // ============== Epilogue: + b2 + residual x, LayerNorm, write ============
  float b2v[16], gv[16], btv[16];
#pragma unroll
  for (int ii = 0; ii < 16; ++ii) {
    const int d = lane * 16 + ii;
    b2v[ii] = b2[p * ND + d];
    gv[ii]  = gamma[d];
    btv[ii] = beta[d];
  }
#pragma unroll 1
  for (int rr = 0; rr < 8; ++rr) {
    const int row = wave + rr * 8;  // each wave handles 8 rows
    const float* xr2 = xblk + (long)row * (NP * ND);
    float vreg[16];
    float s1 = 0.0f, s2 = 0.0f;
#pragma unroll
    for (int q = 0; q < 4; ++q) {
      const int d = lane * 16 + q * 4;
      float4 xv = *(const float4*)(xr2 + d);
      float4 ov = *(const float4*)(res + row * ND + d);
      float t0 = ov.x + b2v[q * 4 + 0] + xv.x;
      float t1 = ov.y + b2v[q * 4 + 1] + xv.y;
      float t2 = ov.z + b2v[q * 4 + 2] + xv.z;
      float t3 = ov.w + b2v[q * 4 + 3] + xv.w;
      vreg[q * 4 + 0] = t0; vreg[q * 4 + 1] = t1;
      vreg[q * 4 + 2] = t2; vreg[q * 4 + 3] = t3;
      s1 += t0 + t1 + t2 + t3;
      s2 += t0 * t0 + t1 * t1 + t2 * t2 + t3 * t3;
    }
    // wave32 butterfly reduction
#pragma unroll
    for (int off = 16; off >= 1; off >>= 1) {
      s1 += __shfl_xor(s1, off, 32);
      s2 += __shfl_xor(s2, off, 32);
    }
    const float mu   = s1 * (1.0f / 512.0f);
    const float var  = s2 * (1.0f / 512.0f) - mu * mu;
    const float rstd = rsqrtf(var + 1e-5f);
    float* orow = out + ((long)(b0 + row) * NP + p) * ND;
#pragma unroll
    for (int q = 0; q < 4; ++q) {
      const int d = lane * 16 + q * 4;
      float4 o4;
      o4.x = (vreg[q * 4 + 0] - mu) * rstd * gv[q * 4 + 0] + btv[q * 4 + 0];
      o4.y = (vreg[q * 4 + 1] - mu) * rstd * gv[q * 4 + 1] + btv[q * 4 + 1];
      o4.z = (vreg[q * 4 + 2] - mu) * rstd * gv[q * 4 + 2] + btv[q * 4 + 2];
      o4.w = (vreg[q * 4 + 3] - mu) * rstd * gv[q * 4 + 3] + btv[q * 4 + 3];
      *(float4*)(orow + d) = o4;
    }
  }
}

extern "C" void kernel_launch(void* const* d_in, const int* in_sizes, int n_in,
                              void* d_out, int out_size, void* d_ws, size_t ws_size,
                              hipStream_t stream) {
  (void)n_in; (void)out_size; (void)d_ws; (void)ws_size;
  const float* x     = (const float*)d_in[0];
  const float* W1    = (const float*)d_in[1];
  const float* b1    = (const float*)d_in[2];
  const float* W2    = (const float*)d_in[3];
  const float* b2    = (const float*)d_in[4];
  const float* gamma = (const float*)d_in[5];
  const float* beta  = (const float*)d_in[6];
  float* out = (float*)d_out;

  const int B = in_sizes[0] / (NP * ND);  // 2048
  dim3 grid(B / BM, NP);                  // (32, 54)
  dim3 block(256);                        // 8 wave32
  hipLaunchKernelGGL(fused_patch_mlp_ln_kernel, grid, block, LDS_BYTES, stream,
                     x, W1, b1, W2, b2, gamma, beta, out);
}
